// GIB_4002909520045
// MI455X (gfx1250) — matrix operand order repaired and checked by reading, not compiled, not run on hardware
//
#include <hip/hip_runtime.h>

// ---------------------------------------------------------------------------
// Problem constants (reference: N=200000, E=3.2M, B=2000, D=300)
// ---------------------------------------------------------------------------
constexpr int D    = 300;   // emb dim
constexpr int D2   = 600;   // hidden dim (2*D)
constexpr int BG   = 2000;  // number of graphs
constexpr int KC1  = 10;    // ceil(300/32)  -> K padded to 320
constexpr int NC1  = 38;    // ceil(600/16)  -> N padded to 608 (even!)
constexpr int KC2  = 19;    // ceil(600/32)  -> K padded to 608
constexpr int NC2  = 19;    // ceil(300/16)  -> N padded to 304
constexpr int PPAD = 320;   // row stride of padded "positive" matrix
constexpr int HPAD = 608;   // row stride of padded hidden matrix

typedef __attribute__((ext_vector_type(16))) __bf16 v16bf;
typedef __attribute__((ext_vector_type(8)))  float  v8f;

union Frag {
    v16bf          v;
    unsigned short u[16];
    uint4          q[2];
};

__device__ __forceinline__ unsigned short f2bf(float f) {
    unsigned int x = __float_as_uint(f);
    x += 0x7fffu + ((x >> 16) & 1u);   // round-to-nearest-even
    return (unsigned short)(x >> 16);
}

__device__ __forceinline__ v16bf zero_frag() {
    Frag fr;
    fr.q[0] = make_uint4(0, 0, 0, 0);
    fr.q[1] = make_uint4(0, 0, 0, 0);
    return fr.v;
}

// A-matrix 16x32 bf16 fragment (wave32 layout, CDNA5 ISA 7.12.2):
// lanes 0-15: M=lane, K = {c0..c0+7, c0+16..c0+23}; lanes 16-31: +8.
// Caller passes col0 = kc*32 + hi*8; row and col0 are 16B aligned.
__device__ __forceinline__ v16bf load_a_frag_vec(const float* __restrict__ row, int col0) {
    const float4 x0 = *(const float4*)(row + col0);
    const float4 x1 = *(const float4*)(row + col0 + 4);
    const float4 x2 = *(const float4*)(row + col0 + 16);
    const float4 x3 = *(const float4*)(row + col0 + 20);
    Frag fr;
    fr.u[0]  = f2bf(x0.x); fr.u[1]  = f2bf(x0.y); fr.u[2]  = f2bf(x0.z); fr.u[3]  = f2bf(x0.w);
    fr.u[4]  = f2bf(x1.x); fr.u[5]  = f2bf(x1.y); fr.u[6]  = f2bf(x1.z); fr.u[7]  = f2bf(x1.w);
    fr.u[8]  = f2bf(x2.x); fr.u[9]  = f2bf(x2.y); fr.u[10] = f2bf(x2.z); fr.u[11] = f2bf(x2.w);
    fr.u[12] = f2bf(x3.x); fr.u[13] = f2bf(x3.y); fr.u[14] = f2bf(x3.z); fr.u[15] = f2bf(x3.w);
    return fr.v;
}

// Guarded variant for the ragged last K chunk (k < KMAX else 0).
template<int KMAX>
__device__ __forceinline__ v16bf load_a_frag_guard(const float* __restrict__ row, int col0) {
    Frag fr;
    #pragma unroll
    for (int h = 0; h < 2; ++h) {
        const int c = col0 + h * 16;
        #pragma unroll
        for (int j = 0; j < 8; ++j) {
            const int k = c + j;
            const float v = (k < KMAX) ? row[k] : 0.0f;
            fr.u[h * 8 + j] = f2bf(v);
        }
    }
    return fr.v;
}

// B fragments pre-swizzled in ws: tile = nc*KC + kc, 1024 B per tile,
// 32 bytes per lane -> two b128 loads.
__device__ __forceinline__ v16bf load_b_frag(const uint4* __restrict__ base, int tile, int lane) {
    Frag fr;
    const uint4* p = base + (size_t)tile * 64 + (size_t)lane * 2;
    fr.q[0] = p[0];
    fr.q[1] = p[1];
    return fr.v;
}

template<int KC>
__device__ __forceinline__ void load_panel(const uint4* __restrict__ base, int nc, int lane,
                                           v16bf b[]) {
    #pragma unroll
    for (int kc = 0; kc < KC; ++kc)
        b[kc] = load_b_frag(base, nc * KC + kc, lane);
}

template<int KC>
__device__ __forceinline__ v8f wmma_chain(const v16bf a[], const v16bf b[]) {
    v8f acc = {0.f, 0.f, 0.f, 0.f, 0.f, 0.f, 0.f, 0.f};
    #pragma unroll
    for (int kc = 0; kc < KC; ++kc)
        acc = __builtin_amdgcn_wmma_f32_16x16x32_bf16(
            false, a[kc], false, b[kc], (short)0, acc, false, false);
    return acc;
}

// ---------------------------------------------------------------------------
// Small prep kernels
// ---------------------------------------------------------------------------
__global__ void zero_kernel(unsigned* flag, unsigned* cnt, int N) {
    const int i = blockIdx.x * blockDim.x + threadIdx.x;
    if (i < N)  flag[i] = 0u;
    if (i < BG) cnt[i]  = 0u;
}

__global__ void scatter_kernel(const int* __restrict__ src, unsigned* __restrict__ flag, int E) {
    const int i = blockIdx.x * blockDim.x + threadIdx.x;
    if (i < E) flag[src[i]] = 1u;   // benign race: everyone stores 1
}

__global__ void hist_kernel(const int* __restrict__ batch, unsigned* __restrict__ cnt, int N) {
    const int i = blockIdx.x * blockDim.x + threadIdx.x;
    if (i < N) atomicAdd(&cnt[batch[i]], 1u);
}

__global__ __launch_bounds__(1024) void scan_kernel(const unsigned* __restrict__ cnt,
                                                    unsigned* __restrict__ off) {
    __shared__ unsigned sd[1024];
    const int t = threadIdx.x;
    const unsigned c0 = (2 * t     < BG) ? cnt[2 * t]     : 0u;
    const unsigned c1 = (2 * t + 1 < BG) ? cnt[2 * t + 1] : 0u;
    const unsigned ts = c0 + c1;
    sd[t] = ts;
    __syncthreads();
    for (int o = 1; o < 1024; o <<= 1) {
        const unsigned v = (t >= o) ? sd[t - o] : 0u;
        __syncthreads();
        sd[t] += v;
        __syncthreads();
    }
    const unsigned excl = sd[t] - ts;
    if (2 * t     < BG) off[2 * t]     = excl;
    if (2 * t + 1 < BG) off[2 * t + 1] = excl + c0;
    if (t == 1023)      off[BG] = sd[1023];
}

__global__ void w2d_kernel(const float* __restrict__ aw2, float* __restrict__ w2d) {
    const int j = blockIdx.x * blockDim.x + threadIdx.x;
    if (j < D2) w2d[j] = aw2[2 * j] - aw2[2 * j + 1];   // softmax over 2 -> sigmoid(l0-l1)
}

// Swizzle a row-major fp32 [K, Ncols] weight into bf16 WMMA B-fragment tiles.
// B layout (32x16 bf16): lanes 0-15 -> N=lane, K = kc*32 + {0..7, 16..23};
// lanes 16-31 -> N=lane-16, K = kc*32 + {8..15, 24..31}.
__global__ void swizzle_kernel(const float* __restrict__ W, unsigned short* __restrict__ out,
                               int K, int Ncols, int KC, int NC) {
    const int gid   = blockIdx.x * blockDim.x + threadIdx.x;
    const int total = KC * NC * 32;
    if (gid >= total) return;
    const int lane  = gid & 31;
    const int tile  = gid >> 5;
    const int kc    = tile % KC;
    const int nc    = tile / KC;
    const int n     = nc * 16 + (lane & 15);
    const int kbase = kc * 32 + ((lane >> 4) << 3);
    unsigned short* o = out + (size_t)gid * 16;
    #pragma unroll
    for (int j = 0; j < 16; ++j) {
        const int k = kbase + j + ((j >> 3) << 3);
        const float v = (k < K && n < Ncols) ? W[(size_t)k * Ncols + n] : 0.0f;
        o[j] = f2bf(v);
    }
}

// ---------------------------------------------------------------------------
// Attention kernel: s = tanh(h@aw1+ab1) . (aw2[:,0]-aw2[:,1]);
// att0 = sigmoid(s + b2d) * valid.  One wave per 16-node tile, 8 waves/block.
// B panels register-double-buffered across the 38 N-tiles.
// ---------------------------------------------------------------------------
__global__ __launch_bounds__(256) void att_kernel(
    const float* __restrict__ h, const uint4* __restrict__ wB1,
    const float* __restrict__ ab1, const float* __restrict__ w2d,
    const float* __restrict__ ab2, const unsigned* __restrict__ flag,
    float* __restrict__ att0, int N)
{
    __shared__ float s_s[8][16];
    const int lane  = threadIdx.x & 31;
    const int wave  = threadIdx.x >> 5;
    const int tbase = blockIdx.x * 128 + wave * 16;
    if (tbase >= N) return;                       // no block-wide sync below: safe
    const int m    = lane & 15;
    const int hi   = lane >> 4;
    const int node = tbase + m;
    const bool rv  = (node < N);
    const float* row = rv ? (h + (size_t)node * D) : h;   // dummy ptr when invalid

    v16bf a[KC1];
    if (rv) {
        #pragma unroll
        for (int kc = 0; kc < KC1 - 1; ++kc)      // chunks 0..8 fully in-bounds
            a[kc] = load_a_frag_vec(row, kc * 32 + hi * 8);
        a[KC1 - 1] = load_a_frag_guard<D>(row, (KC1 - 1) * 32 + hi * 8);
    } else {
        #pragma unroll
        for (int kc = 0; kc < KC1; ++kc) a[kc] = zero_frag();
    }

    float pacc[8];
    #pragma unroll
    for (int r = 0; r < 8; ++r) pacc[r] = 0.0f;

    auto epilogue = [&](int nc, const v8f& acc) {
        const int n = nc * 16 + m;                // this lane's column
        if (n < D2) {
            const float bias = ab1[n];
            const float w    = w2d[n];
            #pragma unroll
            for (int r = 0; r < 8; ++r)
                pacc[r] += tanhf(acc[r] + bias) * w;
        }
    };

    v16bf b0[KC1], b1[KC1];
    load_panel<KC1>(wB1, 0, lane, b0);
    for (int nc = 0; nc < NC1; nc += 2) {         // NC1 even
        load_panel<KC1>(wB1, nc + 1, lane, b1);   // prefetch next panel
        epilogue(nc, wmma_chain<KC1>(a, b0));
        if (nc + 2 < NC1) load_panel<KC1>(wB1, nc + 2, lane, b0);
        epilogue(nc + 1, wmma_chain<KC1>(a, b1));
    }

    // Reduce across the 16 lanes of each half (columns) -> s per node, via LDS bounce.
    #pragma unroll
    for (int r = 0; r < 8; ++r) {
        float v = pacc[r];
        v += __shfl_xor(v, 1, 32);
        v += __shfl_xor(v, 2, 32);
        v += __shfl_xor(v, 4, 32);
        v += __shfl_xor(v, 8, 32);
        if (m == r) s_s[wave][hi * 8 + r] = v;    // node M = r + 8*hi
    }
    // Same-wave DS ops are in order; compiler waits DScnt before dependent use.
    if (lane < 16) {
        const int nd = tbase + lane;
        if (nd < N) {
            const float b2d = ab2[0] - ab2[1];
            const float s   = s_s[wave][lane];
            const float a0  = 1.0f / (1.0f + __expf(-(s + b2d)));
            att0[nd] = flag[nd] ? a0 : 0.0f;
        }
    }
}

// ---------------------------------------------------------------------------
// Group feature: one block per graph (batch is sorted -> contiguous segment).
// No atomics; each thread owns one embedding column.  Writes clipped+padded.
// ---------------------------------------------------------------------------
__global__ __launch_bounds__(320) void gf_kernel(
    const float* __restrict__ h, const float* __restrict__ att0,
    const unsigned* __restrict__ off, float* __restrict__ pos)
{
    const int g = blockIdx.x;
    const int d = threadIdx.x;
    const unsigned s0 = off[g];
    const unsigned s1 = off[g + 1];
    float acc = 0.0f;
    if (d < D) {
        for (unsigned n = s0; n < s1; ++n)
            acc = fmaf(att0[n], h[(size_t)n * D + d], acc);
        acc = fminf(fmaxf(acc, -100.0f), 100.0f);
    }
    pos[(size_t)g * PPAD + d] = acc;              // d in [300,320) -> zero pad
}

// ---------------------------------------------------------------------------
// cls layer 1: relu(positive @ cw1 + cb1) -> hid (padded fp32)
// ---------------------------------------------------------------------------
__global__ __launch_bounds__(256) void cls1_kernel(
    const float* __restrict__ pos, const uint4* __restrict__ wC1,
    const float* __restrict__ cb1, float* __restrict__ hid)
{
    const int lane  = threadIdx.x & 31;
    const int wave  = threadIdx.x >> 5;
    const int rbase = blockIdx.x * 128 + wave * 16;
    if (rbase >= BG) return;
    const int m  = lane & 15;
    const int hi = lane >> 4;
    const int rr = rbase + m;
    const bool rv = (rr < BG);
    const float* row = rv ? (pos + (size_t)rr * PPAD) : pos;

    v16bf a[KC1];
    if (rv) {
        #pragma unroll
        for (int kc = 0; kc < KC1; ++kc)          // pos padded: all in-bounds
            a[kc] = load_a_frag_vec(row, kc * 32 + hi * 8);
    } else {
        #pragma unroll
        for (int kc = 0; kc < KC1; ++kc) a[kc] = zero_frag();
    }

    auto epilogue = [&](int nc, const v8f& acc) {
        const int n = nc * 16 + m;
        const float bias = (n < D2) ? cb1[n] : 0.0f;
        #pragma unroll
        for (int r = 0; r < 8; ++r) {
            const int orow = rbase + r + 8 * hi;  // D-tile: VGPR r, half hi
            if (orow < BG) {
                const float v = (n < D2) ? fmaxf(acc[r] + bias, 0.0f) : 0.0f;
                hid[(size_t)orow * HPAD + n] = v; // pad cols 600..607 zeroed here
            }
        }
    };

    v16bf b0[KC1], b1[KC1];
    load_panel<KC1>(wC1, 0, lane, b0);
    for (int nc = 0; nc < NC1; nc += 2) {
        load_panel<KC1>(wC1, nc + 1, lane, b1);
        epilogue(nc, wmma_chain<KC1>(a, b0));
        if (nc + 2 < NC1) load_panel<KC1>(wC1, nc + 2, lane, b0);
        epilogue(nc + 1, wmma_chain<KC1>(a, b1));
    }
}

// ---------------------------------------------------------------------------
// cls layer 2: relu(hid @ cw2 + cb2) -> pred (d_out)
// ---------------------------------------------------------------------------
__global__ __launch_bounds__(256) void cls2_kernel(
    const float* __restrict__ hid, const uint4* __restrict__ wC2,
    const float* __restrict__ cb2, float* __restrict__ out)
{
    const int lane  = threadIdx.x & 31;
    const int wave  = threadIdx.x >> 5;
    const int rbase = blockIdx.x * 128 + wave * 16;
    if (rbase >= BG) return;
    const int m  = lane & 15;
    const int hi = lane >> 4;
    const int rr = rbase + m;
    const bool rv = (rr < BG);
    const float* row = rv ? (hid + (size_t)rr * HPAD) : hid;

    v16bf a[KC2];
    if (rv) {
        #pragma unroll
        for (int kc = 0; kc < KC2; ++kc)          // hid padded: all in-bounds
            a[kc] = load_a_frag_vec(row, kc * 32 + hi * 8);
    } else {
        #pragma unroll
        for (int kc = 0; kc < KC2; ++kc) a[kc] = zero_frag();
    }

    for (int nc = 0; nc < NC2; ++nc) {
        v16bf b[KC2];
        load_panel<KC2>(wC2, nc, lane, b);        // panel hoist: 1 wait / 19 WMMAs
        const v8f acc = wmma_chain<KC2>(a, b);
        const int n = nc * 16 + m;
        if (n < D) {
            const float bias = cb2[n];
            #pragma unroll
            for (int r = 0; r < 8; ++r) {
                const int orow = rbase + r + 8 * hi;
                if (orow < BG)
                    out[(size_t)orow * D + n] = fmaxf(acc[r] + bias, 0.0f);
            }
        }
    }
}

// ---------------------------------------------------------------------------
// Launch
// ---------------------------------------------------------------------------
extern "C" void kernel_launch(void* const* d_in, const int* in_sizes, int n_in,
                              void* d_out, int out_size, void* d_ws, size_t ws_size,
                              hipStream_t stream) {
    const float* h    = (const float*)d_in[0];
    const int*   batch= (const int*)  d_in[1];
    const int*   eidx = (const int*)  d_in[2];   // [2, E]; row 0 = sources
    const float* aw1  = (const float*)d_in[3];
    const float* ab1  = (const float*)d_in[4];
    const float* aw2  = (const float*)d_in[5];
    const float* ab2  = (const float*)d_in[6];
    const float* cw1  = (const float*)d_in[7];
    const float* cb1  = (const float*)d_in[8];
    const float* cw2  = (const float*)d_in[9];
    const float* cb2  = (const float*)d_in[10];
    float* out = (float*)d_out;
    const int N = in_sizes[1];
    const int E = in_sizes[2] / 2;
    (void)n_in; (void)out_size; (void)ws_size;

    // carve workspace (256B aligned slices)
    char* p = (char*)d_ws;
    auto take = [&](size_t bytes) -> char* {
        char* r = p;
        p += (bytes + 255) & ~(size_t)255;
        return r;
    };
    unsigned* flag = (unsigned*)take((size_t)N * 4);
    unsigned* cnt  = (unsigned*)take((size_t)BG * 4);
    unsigned* off  = (unsigned*)take((size_t)(BG + 1) * 4);
    float*    w2d  = (float*)   take((size_t)D2 * 4);
    uint4*    wB1  = (uint4*)   take((size_t)KC1 * NC1 * 1024);
    uint4*    wC1  = (uint4*)   take((size_t)KC1 * NC1 * 1024);
    uint4*    wC2  = (uint4*)   take((size_t)KC2 * NC2 * 1024);
    float*    att0 = (float*)   take((size_t)N * 4);
    float*    pos  = (float*)   take((size_t)BG * PPAD * 4);
    float*    hid  = (float*)   take((size_t)BG * HPAD * 4);

    auto nb = [](int work, int block) { return (work + block - 1) / block; };

    zero_kernel   <<<nb(N, 256), 256, 0, stream>>>(flag, cnt, N);
    scatter_kernel<<<nb(E, 256), 256, 0, stream>>>(eidx, flag, E);
    hist_kernel   <<<nb(N, 256), 256, 0, stream>>>(batch, cnt, N);
    scan_kernel   <<<1, 1024, 0, stream>>>(cnt, off);
    w2d_kernel    <<<nb(D2, 256), 256, 0, stream>>>(aw2, w2d);

    swizzle_kernel<<<nb(KC1 * NC1 * 32, 256), 256, 0, stream>>>(
        aw1, (unsigned short*)wB1, D,  D2, KC1, NC1);
    swizzle_kernel<<<nb(KC1 * NC1 * 32, 256), 256, 0, stream>>>(
        cw1, (unsigned short*)wC1, D,  D2, KC1, NC1);
    swizzle_kernel<<<nb(KC2 * NC2 * 32, 256), 256, 0, stream>>>(
        cw2, (unsigned short*)wC2, D2, D,  KC2, NC2);

    att_kernel <<<nb(N, 128), 256, 0, stream>>>(h, wB1, ab1, w2d, ab2, flag, att0, N);
    gf_kernel  <<<BG, 320, 0, stream>>>(h, att0, off, pos);
    cls1_kernel<<<nb(BG, 128), 256, 0, stream>>>(pos, wC1, cb1, hid);
    cls2_kernel<<<nb(BG, 128), 256, 0, stream>>>(hid, wC2, cb2, out);
}
